// RoMoE_8658654069080
// MI455X (gfx1250) — matrix operand order repaired and checked by reading, not compiled, not run on hardware
//
#include <hip/hip_runtime.h>

typedef __attribute__((ext_vector_type(16))) _Float16 v16h;
typedef __attribute__((ext_vector_type(8)))  _Float16 v8h;
typedef __attribute__((ext_vector_type(4)))  _Float16 v4h;
typedef __attribute__((ext_vector_type(8)))  float    v8f;

#define D_IN  1024
#define D_HID 2048
#define D_OUT 1024
#define BATCH 4096
#define NE    8      // active experts (first half of 16)
#define BM    32     // rows per block

// ---------------------------------------------------------------------------
// fp32 -> f16 conversion of the three input matrices into ws (contiguous [3][B][D_IN])
// ---------------------------------------------------------------------------
__global__ __launch_bounds__(256)
void cvt_inputs(const float* __restrict__ x, const float* __restrict__ y,
                const float* __restrict__ z, _Float16* __restrict__ o) {
    const int c = blockIdx.y;
    const float* src = (c == 0) ? x : (c == 1) ? y : z;
    size_t idx = ((size_t)blockIdx.x * 256 + threadIdx.x) * 4;
    float4 v = *(const float4*)(src + idx);
    v4h h;
    h[0] = (_Float16)v.x; h[1] = (_Float16)v.y;
    h[2] = (_Float16)v.z; h[3] = (_Float16)v.w;
    *(v4h*)(o + (size_t)c * BATCH * D_IN + idx) = h;
}

// ---------------------------------------------------------------------------
// Transpose-convert W[e][K][N] fp32 -> WT[e][N][K] f16 (first 8 experts).
// ---------------------------------------------------------------------------
__global__ __launch_bounds__(256)
void transcvt(const float* __restrict__ in, _Float16* __restrict__ outp, int K, int N) {
    __shared__ float tile[32][33];
    const int e  = blockIdx.z;
    const int k0 = blockIdx.x * 32;
    const int n0 = blockIdx.y * 32;
    const int tx = threadIdx.x, ty = threadIdx.y;   // 32 x 8
    const float*    src = in   + (size_t)e * K * N;
    _Float16*       dst = outp + (size_t)e * K * N;
    #pragma unroll
    for (int i = 0; i < 32; i += 8)
        tile[ty + i][tx] = src[(size_t)(k0 + ty + i) * N + (n0 + tx)];
    __syncthreads();
    #pragma unroll
    for (int i = 0; i < 32; i += 8)
        dst[(size_t)(n0 + ty + i) * K + (k0 + tx)] = (_Float16)tile[tx][ty + i];
}

// ---------------------------------------------------------------------------
// gates[c][b][e] = softmax_e( inp_c[b,:] @ Wg_c[:,e] )
// ---------------------------------------------------------------------------
__global__ __launch_bounds__(256)
void gates_k(const float* __restrict__ x, const float* __restrict__ y,
             const float* __restrict__ z, const float* __restrict__ Wg1,
             const float* __restrict__ Wg2, const float* __restrict__ Wg3,
             float* __restrict__ g) {
    __shared__ float sW[D_IN * NE];
    const int c = blockIdx.y;
    const float* inp = (c == 0) ? x : (c == 1) ? y : z;
    const float* Wg  = (c == 0) ? Wg1 : (c == 1) ? Wg2 : Wg3;
    for (int i = threadIdx.x; i < D_IN * NE; i += 256) sW[i] = Wg[i];
    __syncthreads();
    const int r = blockIdx.x * 256 + threadIdx.x;
    const float* row = inp + (size_t)r * D_IN;
    float a[NE];
    #pragma unroll
    for (int e = 0; e < NE; ++e) a[e] = 0.f;
    for (int k = 0; k < D_IN; ++k) {
        const float xv = row[k];
        #pragma unroll
        for (int e = 0; e < NE; ++e) a[e] += xv * sW[k * NE + e];
    }
    float mx = a[0];
    #pragma unroll
    for (int e = 1; e < NE; ++e) mx = fmaxf(mx, a[e]);
    float s = 0.f;
    #pragma unroll
    for (int e = 0; e < NE; ++e) { a[e] = expf(a[e] - mx); s += a[e]; }
    const float inv = 1.f / s;
    float* go = g + ((size_t)c * BATCH + r) * NE;
    #pragma unroll
    for (int e = 0; e < NE; ++e) go[e] = a[e] * inv;
}

// ---------------------------------------------------------------------------
// Fused MoE main kernel (see round-1 comments for fragment layouts).
// k-outer / n-inner loops: A fragments loaded once per k-step and reused
// across 4 (GEMM1) / 8 (GEMM2) n-tiles -> 4-8x less LDS read traffic.
// Input tile staged LDS-side with CDNA5 async copies (ASYNCcnt).
// ---------------------------------------------------------------------------
__global__ __launch_bounds__(256)
void moe_main(const _Float16* __restrict__ xh,    // [3][B][D_IN] f16
              const _Float16* __restrict__ w1t,   // [8][D_HID][D_IN] f16
              const _Float16* __restrict__ w2t,   // [8][D_OUT][D_HID] f16
              const float* __restrict__ b1,       // [16][D_HID] f32
              const float* __restrict__ b2,       // [16][D_OUT] f32
              const float* __restrict__ gts,      // [3][B][8] f32
              float* __restrict__ out)            // [B][3*D_OUT] f32
{
    extern __shared__ char smem[];
    _Float16* sIn = (_Float16*)smem;                          // 32*1024 f16 = 64 KB (offset 0)
    _Float16* sH  = (_Float16*)(smem + 65536);                // 32*2048 f16 = 128 KB
    float*    sG  = (float*)(smem + 65536 + 131072);          // 32*8 f32

    const int c     = blockIdx.y;
    const int m0    = blockIdx.x * BM;
    const int tid   = threadIdx.x;
    const int wave  = tid >> 5;
    const int lane  = tid & 31;
    const int lhalf = lane >> 4;     // 0/1
    const int l16   = lane & 15;

    // ---- stage input tile via async global->LDS copies (dynamic LDS starts at offset 0) ----
    const _Float16* inp = xh + ((size_t)c * BATCH + m0) * D_IN;
    for (int i = tid; i < BM * D_IN / 8; i += 256) {
        unsigned lds_off = (unsigned)(i * 16);                       // byte offset of v8h #i
        unsigned long long ga = (unsigned long long)(inp) + (unsigned long long)i * 16;
        asm volatile("global_load_async_to_lds_b128 %0, %1, off"
                     :: "v"(lds_off), "v"(ga) : "memory");
    }
    for (int i = tid; i < BM * NE; i += 256)
        sG[i] = gts[((size_t)c * BATCH + m0) * NE + i];
    asm volatile("s_wait_asynccnt 0x0" ::: "memory");
    __syncthreads();

    // persistent output accumulators: 8 n-tiles x 2 m-tiles
    v8f acc[16];
    #pragma unroll
    for (int i = 0; i < 16; ++i) acc[i] = (v8f)0.0f;

    for (int e = 0; e < NE; ++e) {
        // per-lane gate values for the two m-tiles (m = r + 8*lhalf [+16])
        float g0[8], g1[8];
        #pragma unroll
        for (int r = 0; r < 8; ++r) {
            g0[r] = sG[(r + 8 * lhalf) * NE + e];
            g1[r] = sG[(16 + r + 8 * lhalf) * NE + e];
        }

        // ---------------- GEMM1: 32 x 2048, K = 1024 ----------------
        // wave handles 16 n-tiles, processed as 4 chunks of 4 (k-outer inside chunk)
        const _Float16* w1e = w1t + (size_t)e * D_HID * D_IN;
        const _Float16* a0i = sIn + (size_t)l16 * D_IN + lhalf * 8;
        const _Float16* a1i = sIn + (size_t)(16 + l16) * D_IN + lhalf * 8;
        for (int ntc = 0; ntc < 4; ++ntc) {
            int ncol[4];
            const _Float16* bp[4];
            #pragma unroll
            for (int q = 0; q < 4; ++q) {
                ncol[q] = (wave + (ntc * 4 + q) * 8) * 16 + l16;
                bp[q]   = w1e + (size_t)ncol[q] * D_IN + lhalf * 8;
                __builtin_prefetch(bp[q], 0, 0);
            }
            v8f h[4][2];
            #pragma unroll
            for (int q = 0; q < 4; ++q) { h[q][0] = (v8f)0.0f; h[q][1] = (v8f)0.0f; }
            #pragma unroll 2
            for (int ks = 0; ks < D_IN / 32; ++ks) {
                const int k0 = ks * 32;
                v16h af0, af1;
                *(v8h*)&af0       = *(const v8h*)(a0i + k0);
                *((v8h*)&af0 + 1) = *(const v8h*)(a0i + k0 + 16);
                *(v8h*)&af1       = *(const v8h*)(a1i + k0);
                *((v8h*)&af1 + 1) = *(const v8h*)(a1i + k0 + 16);
                #pragma unroll
                for (int q = 0; q < 4; ++q) {
                    v16h bf;
                    *(v8h*)&bf       = *(const v8h*)(bp[q] + k0);
                    *((v8h*)&bf + 1) = *(const v8h*)(bp[q] + k0 + 16);
                    h[q][0] = __builtin_amdgcn_wmma_f32_16x16x32_f16(false, af0, false, bf, (short)0, h[q][0], false, false);
                    h[q][1] = __builtin_amdgcn_wmma_f32_16x16x32_f16(false, af1, false, bf, (short)0, h[q][1], false, false);
                }
            }
            // epilogue: +b1, relu, *gate, f16 -> sH
            #pragma unroll
            for (int q = 0; q < 4; ++q) {
                const float b1v = b1[(size_t)e * D_HID + ncol[q]];
                #pragma unroll
                for (int r = 0; r < 8; ++r) {
                    float v0 = h[q][0][r] + b1v; v0 = v0 > 0.f ? v0 : 0.f; v0 *= g0[r];
                    float v1 = h[q][1][r] + b1v; v1 = v1 > 0.f ? v1 : 0.f; v1 *= g1[r];
                    sH[(size_t)(r + 8 * lhalf) * D_HID + ncol[q]]      = (_Float16)v0;
                    sH[(size_t)(16 + r + 8 * lhalf) * D_HID + ncol[q]] = (_Float16)v1;
                }
            }
        }
        __syncthreads();

        // ---------------- GEMM2: 32 x 1024, K = 2048 (accumulate over experts) ----------------
        const _Float16* w2e = w2t + (size_t)e * D_OUT * D_HID;
        const _Float16* bp2[8];
        #pragma unroll
        for (int j = 0; j < 8; ++j) {
            const int ocol = (wave + j * 8) * 16 + l16;
            bp2[j] = w2e + (size_t)ocol * D_HID + lhalf * 8;
            __builtin_prefetch(bp2[j], 0, 0);
        }
        const _Float16* a0h = sH + (size_t)l16 * D_HID + lhalf * 8;
        const _Float16* a1h = sH + (size_t)(16 + l16) * D_HID + lhalf * 8;
        #pragma unroll 1
        for (int ks = 0; ks < D_HID / 32; ++ks) {
            const int k0 = ks * 32;
            v16h af0, af1;
            *(v8h*)&af0       = *(const v8h*)(a0h + k0);
            *((v8h*)&af0 + 1) = *(const v8h*)(a0h + k0 + 16);
            *(v8h*)&af1       = *(const v8h*)(a1h + k0);
            *((v8h*)&af1 + 1) = *(const v8h*)(a1h + k0 + 16);
            #pragma unroll
            for (int j = 0; j < 8; ++j) {
                v16h bf;
                *(v8h*)&bf       = *(const v8h*)(bp2[j] + k0);
                *((v8h*)&bf + 1) = *(const v8h*)(bp2[j] + k0 + 16);
                acc[2 * j]     = __builtin_amdgcn_wmma_f32_16x16x32_f16(false, af0, false, bf, (short)0, acc[2 * j],     false, false);
                acc[2 * j + 1] = __builtin_amdgcn_wmma_f32_16x16x32_f16(false, af1, false, bf, (short)0, acc[2 * j + 1], false, false);
            }
        }
        __syncthreads();   // before next expert overwrites sH
    }

    // ---------------- epilogue: + sum_e g*b2, write concat output ----------------
    float* outp = out + (size_t)m0 * (3 * D_OUT) + (size_t)c * D_OUT;
    #pragma unroll
    for (int j = 0; j < 8; ++j) {
        const int ocol = (wave + j * 8) * 16 + l16;
        float b2v[NE];
        #pragma unroll
        for (int e = 0; e < NE; ++e) b2v[e] = b2[(size_t)e * D_OUT + ocol];
        #pragma unroll
        for (int mt = 0; mt < 2; ++mt) {
            const v8f a = acc[2 * j + mt];
            #pragma unroll
            for (int r = 0; r < 8; ++r) {
                const int m = mt * 16 + 8 * lhalf + r;
                float bias = 0.f;
                #pragma unroll
                for (int e = 0; e < NE; ++e) bias += sG[m * NE + e] * b2v[e];
                outp[(size_t)m * (3 * D_OUT) + ocol] = a[r] + bias;
            }
        }
    }
}

// ---------------------------------------------------------------------------
extern "C" void kernel_launch(void* const* d_in, const int* in_sizes, int n_in,
                              void* d_out, int out_size, void* d_ws, size_t ws_size,
                              hipStream_t stream) {
    const float* x   = (const float*)d_in[0];
    const float* y   = (const float*)d_in[1];
    const float* z   = (const float*)d_in[2];
    const float* W1  = (const float*)d_in[3];
    const float* b1  = (const float*)d_in[4];
    const float* W2  = (const float*)d_in[5];
    const float* b2  = (const float*)d_in[6];
    const float* Wg1 = (const float*)d_in[7];
    const float* Wg2 = (const float*)d_in[8];
    const float* Wg3 = (const float*)d_in[9];
    float* out = (float*)d_out;

    char* ws = (char*)d_ws;
    _Float16* xh  = (_Float16*)(ws);                 // 3*4096*1024*2  = 25165824 B
    _Float16* w1t = (_Float16*)(ws + 25165824);      // 8*2048*1024*2  = 33554432 B
    _Float16* w2t = (_Float16*)(ws + 58720256);      // 8*1024*2048*2  = 33554432 B
    float*    gts = (float*)(ws + 92274688);         // 3*4096*8*4     =   393216 B

    cvt_inputs<<<dim3(BATCH * D_IN / 1024, 3), 256, 0, stream>>>(x, y, z, xh);
    transcvt<<<dim3(D_IN / 32, D_HID / 32, NE), dim3(32, 8), 0, stream>>>(W1, w1t, D_IN, D_HID);
    transcvt<<<dim3(D_HID / 32, D_OUT / 32, NE), dim3(32, 8), 0, stream>>>(W2, w2t, D_HID, D_OUT);
    gates_k<<<dim3(BATCH / 256, 3), 256, 0, stream>>>(x, y, z, Wg1, Wg2, Wg3, gts);

    const size_t shmem = 65536 + 131072 + BM * NE * sizeof(float);   // 197,632 B
    hipFuncSetAttribute((const void*)moe_main, hipFuncAttributeMaxDynamicSharedMemorySize, (int)shmem);
    moe_main<<<dim3(BATCH / BM, 3), 256, shmem, stream>>>(xh, w1t, w2t, b1, b2, gts, out);
}
